// ConvLayer_50869592655536
// MI455X (gfx1250) — matrix operand order repaired and checked by reading, not compiled, not run on hardware
//
#include <hip/hip_runtime.h>
#include <hip/hip_bf16.h>

// ---------------------------------------------------------------------------
// Problem constants (from the reference)
// ---------------------------------------------------------------------------
#define NN 65536      // rows per subgraph type
#define MM 262144     // gather pairs per map
#define HH 128        // hidden
#define TT 4          // path types

typedef __attribute__((ext_vector_type(16))) __bf16 v16bf;
typedef __attribute__((ext_vector_type(8)))  __bf16 v8bf;
typedef __attribute__((ext_vector_type(8)))  float  v8f;

union B16x16 { v16bf v; v8bf h[2]; };

static __device__ __forceinline__ __bf16 f2bf(float f) {
  unsigned u = __builtin_bit_cast(unsigned, f);
  unsigned r = (u + 0x7FFFu + ((u >> 16) & 1u)) >> 16;   // RNE
  unsigned short s = (unsigned short)r;
  return __builtin_bit_cast(__bf16, s);
}

// pack 4 f32 -> 4 bf16, single 8-byte store
static __device__ __forceinline__ void store_bf4(__bf16* p, float4 v) {
  union { __bf16 b[4]; uint2 u; } o;
  o.b[0] = f2bf(v.x); o.b[1] = f2bf(v.y); o.b[2] = f2bf(v.z); o.b[3] = f2bf(v.w);
  *(uint2*)p = o.u;
}

// ---------------------------------------------------------------------------
// Vectorized utility kernels (all sizes are multiples of 4; buffers 16B aligned)
// ---------------------------------------------------------------------------
__global__ void zero_f32_v4(float4* __restrict__ p, int n4) {
  int i = blockIdx.x * blockDim.x + threadIdx.x;
  if (i < n4) p[i] = (float4){0.f, 0.f, 0.f, 0.f};
}

__global__ void accum_add_v4(float4* __restrict__ y, const float4* __restrict__ x, int n4) {
  int i = blockIdx.x * blockDim.x + threadIdx.x;
  if (i >= n4) return;
  float4 a = y[i], b = x[i];
  y[i] = (float4){a.x + b.x, a.y + b.y, a.z + b.z, a.w + b.w};
}

__global__ void cvt_bf16_v4(const float4* __restrict__ x, __bf16* __restrict__ y, int n4) {
  int i = blockIdx.x * blockDim.x + threadIdx.x;
  if (i < n4) store_bf4(y + (size_t)i * 4, x[i]);
}

// dst[r*ldd + c..c+3] = src[r*lds + c..c+3]
__global__ void copy_cols_v4(float* __restrict__ dst, int ldd,
                             const float* __restrict__ src, int lds_,
                             int nrows, int C) {
  int cq = C >> 2;
  int t = blockIdx.x * blockDim.x + threadIdx.x;
  if (t >= nrows * cq) return;
  int r = t / cq, c = (t - r * cq) * 4;
  *(float4*)(dst + (size_t)r * ldd + c) = *(const float4*)(src + (size_t)r * lds_ + c);
}

// W[Kin,Kout] f32 -> Wt[Kout,Kin] bf16  (so B columns are contiguous K-runs)
__global__ void transpose_w_bf16(const float* __restrict__ W, __bf16* __restrict__ Wt,
                                 int Kin, int Kout) {
  int t = blockIdx.x * blockDim.x + threadIdx.x;
  if (t >= Kin * Kout) return;
  int k = t / Kout, n = t % Kout;
  Wt[(size_t)n * Kin + k] = f2bf(W[t]);
}

// ---------------------------------------------------------------------------
// Gather (segment_sum + count): out[N, 2C] = [sum || sum/max(count,1)]
// ---------------------------------------------------------------------------
__global__ void gather_count(const int* __restrict__ seg, float* __restrict__ cnt, int M) {
  int e = blockIdx.x * blockDim.x + threadIdx.x;
  if (e < M) atomicAdd(&cnt[seg[e]], 1.0f);
}

// one float4 source read + 4 atomic adds per thread
__global__ void gather_scatter_v4(const float* __restrict__ src,
                                  const int* __restrict__ srcIdx,
                                  const int* __restrict__ seg,
                                  float* __restrict__ out, int M, int C) {
  int cq = C >> 2;
  int t = blockIdx.x * blockDim.x + threadIdx.x;
  if (t >= M * cq) return;
  int e = t / cq, c = (t - e * cq) * 4;
  float4 v = *(const float4*)(src + (size_t)srcIdx[e] * C + c);
  float* o = out + (size_t)seg[e] * (2 * C) + c;
  atomicAdd(o + 0, v.x);
  atomicAdd(o + 1, v.y);
  atomicAdd(o + 2, v.z);
  atomicAdd(o + 3, v.w);
}

// finalize mean half; optionally emit a bf16 copy of the whole [N,2C] row block
__global__ void gather_fin_v4(float* __restrict__ out, __bf16* __restrict__ outbf,
                              const float* __restrict__ cnt, int N, int C) {
  int cq = C >> 2;
  int t = blockIdx.x * blockDim.x + threadIdx.x;
  if (t >= N * cq) return;
  int n = t / cq, c = (t - n * cq) * 4;
  float inv = 1.0f / fmaxf(cnt[n], 1.0f);
  float4 s = *(const float4*)(out + (size_t)n * (2 * C) + c);
  float4 m = (float4){s.x * inv, s.y * inv, s.z * inv, s.w * inv};
  *(float4*)(out + (size_t)n * (2 * C) + C + c) = m;
  if (outbf) {
    store_bf4(outbf + (size_t)n * (2 * C) + c, s);
    store_bf4(outbf + (size_t)n * (2 * C) + C + c, m);
  }
}

// ---------------------------------------------------------------------------
// BatchNorm over rows: stats (sum, sumsq), then normalize + ReLU (+ residual),
// with optional f32 and/or bf16 outputs (bf16 feeds the next WMMA GEMM).
// ---------------------------------------------------------------------------
__global__ void bn_stats(const float* __restrict__ x, float* __restrict__ stats,
                         int nrows, int C) {
  int c = blockIdx.x * blockDim.x + threadIdx.x;
  if (c >= C) return;
  int rowsPer = (nrows + gridDim.y - 1) / gridDim.y;
  int r0 = blockIdx.y * rowsPer;
  int r1 = r0 + rowsPer; if (r1 > nrows) r1 = nrows;
  float s = 0.f, s2 = 0.f;
  for (int r = r0; r < r1; ++r) {
    float v = x[(size_t)r * C + c];
    s += v; s2 += v * v;
  }
  atomicAdd(&stats[c], s);
  atomicAdd(&stats[C + c], s2);
}

__global__ void bn_apply_relu_v4(const float* __restrict__ x, const float* __restrict__ stats,
                                 const float* __restrict__ gb,
                                 float* __restrict__ y, __bf16* __restrict__ ybf,
                                 const float* __restrict__ res, int nrows, int C) {
  int total4 = (nrows * C) >> 2;
  int i = blockIdx.x * blockDim.x + threadIdx.x;
  if (i >= total4) return;
  size_t base = (size_t)i * 4;
  int c = (int)(base % C);
  float inv_n = 1.0f / (float)nrows;
  float4 xv = *(const float4*)(x + base);
  float4 sm = *(const float4*)(stats + c);        // channel sums
  float4 sq = *(const float4*)(stats + C + c);    // channel sumsq
  float4 gg = *(const float4*)(gb + c);
  float4 bb = *(const float4*)(gb + C + c);
  float4 o;
  {
    float m = sm.x * inv_n, v = sq.x * inv_n - m * m;
    o.x = fmaxf((xv.x - m) * rsqrtf(v + 1e-5f) * gg.x + bb.x, 0.f);
    m = sm.y * inv_n; v = sq.y * inv_n - m * m;
    o.y = fmaxf((xv.y - m) * rsqrtf(v + 1e-5f) * gg.y + bb.y, 0.f);
    m = sm.z * inv_n; v = sq.z * inv_n - m * m;
    o.z = fmaxf((xv.z - m) * rsqrtf(v + 1e-5f) * gg.z + bb.z, 0.f);
    m = sm.w * inv_n; v = sq.w * inv_n - m * m;
    o.w = fmaxf((xv.w - m) * rsqrtf(v + 1e-5f) * gg.w + bb.w, 0.f);
  }
  if (res) {
    float4 rv = *(const float4*)(res + base);
    o.x += rv.x; o.y += rv.y; o.z += rv.z; o.w += rv.w;
  }
  if (y)   *(float4*)(y + base) = o;
  if (ybf) store_bf4(ybf + base, o);
}

// ---------------------------------------------------------------------------
// bf16 WMMA GEMM:  Out[N,Kout] = concat(A0|A1|A2)[N,Kin](bf16) @ W[Kin,Kout]
//   Wt = W^T as bf16 [Kout, Kin].
//   Block: 256 threads = 8 waves; wave tile = 32 rows x 128 cols
//     -> 16x v_wmma_f32_16x16x32_bf16 per K-step.
//   B chunk (32K x 128N bf16, 8 KB) staged in LDS once per block per K-step
//   (row pitch 40 bf16 = 20 dwords: conflict-free 16-lane ds_load_b128).
//   All 8 B tiles are preloaded into distinct registers before the WMMA burst
//   so ds_load latency is covered with partial DScnt waits instead of
//   serializing each WMMA pair behind s_wait_dscnt 0.
// ---------------------------------------------------------------------------
#define BPITCH 40   // bf16 elements per staged B row (32 data + 8 pad)

__global__ void __launch_bounds__(256)
gemm_bf16_wmma(const __bf16* __restrict__ A0, int c0,
               const __bf16* __restrict__ A1, int c1,
               const __bf16* __restrict__ A2, int c2,
               const __bf16* __restrict__ Wt, int Kin, int Kout,
               float* __restrict__ Out, __bf16* __restrict__ Outbf, int ldc) {
  __shared__ __align__(16) unsigned short Bs[128 * BPITCH];

  const int tid  = threadIdx.x;
  const int lane = tid & 31;
  const int wave = tid >> 5;
  const int row0 = (blockIdx.x * 8 + wave) * 32;   // 32-row strip per wave
  const int col0 = blockIdx.y * 128;               // 128-col span per block
  const int hf   = lane >> 4;
  const int mr   = lane & 15;

  // staging role: 2 threads per B row (each moves 16 bf16 = 32 B)
  const int srow  = tid >> 1;        // 0..127
  const int spart = tid & 1;         // 0..1
  const __bf16* wrow = Wt + (size_t)(col0 + srow) * Kin + spart * 16;
  unsigned short* sdst = Bs + srow * BPITCH + spart * 16;

  v8f acc[2][8];
  #pragma unroll
  for (int g = 0; g < 2; ++g)
    #pragma unroll
    for (int t = 0; t < 8; ++t)
      acc[g][t] = (v8f){0.f,0.f,0.f,0.f,0.f,0.f,0.f,0.f};

  const int r0 = row0 + mr;          // A rows for group 0
  const int r1 = row0 + 16 + mr;     // A rows for group 1

  for (int k0 = 0; k0 < Kin; k0 += 32) {
    // ---- select concat segment (uniform: segment widths are multiples of 32)
    const __bf16* Aseg; int segw, kloc;
    if (k0 < c0)            { Aseg = A0; segw = c0; kloc = k0; }
    else if (k0 < c0 + c1)  { Aseg = A1; segw = c1; kloc = k0 - c0; }
    else                    { Aseg = A2; segw = c2; kloc = k0 - c0 - c1; }
    const __bf16* ap0 = Aseg + (size_t)r0 * segw + kloc;
    const __bf16* ap1 = Aseg + (size_t)r1 * segw + kloc;

    // ---- A tiles: two 16x32 bf16 tiles, 2x b128 each (contiguous runs)
    B16x16 a0, a1;
    a0.h[0] = *(const v8bf*)(ap0 + 8 * hf);
    a0.h[1] = *(const v8bf*)(ap0 + 16 + 8 * hf);
    a1.h[0] = *(const v8bf*)(ap1 + 8 * hf);
    a1.h[1] = *(const v8bf*)(ap1 + 16 + 8 * hf);

    // ---- stage B chunk to LDS (2x b128 per thread), prefetch next chunk
    {
      const uint4* s = (const uint4*)(wrow + k0);
      uint4 x0 = s[0];
      uint4 x1 = s[1];
      if (k0 + 32 < Kin) __builtin_prefetch(wrow + k0 + 32, 0, 1);
      ((uint4*)sdst)[0] = x0;
      ((uint4*)sdst)[1] = x1;
    }
    __syncthreads();

    // ---- preload all 8 B tiles into distinct registers (16x ds_load_b128)
    B16x16 b[8];
    #pragma unroll
    for (int t = 0; t < 8; ++t) {
      const v8bf* bp = (const v8bf*)(Bs + (t * 16 + mr) * BPITCH + 8 * hf);
      b[t].h[0] = bp[0];
      b[t].h[1] = bp[2];
    }
    // ---- 16 WMMA burst
    #pragma unroll
    for (int t = 0; t < 8; ++t) {
      acc[0][t] = __builtin_amdgcn_wmma_f32_16x16x32_bf16(
          false, a0.v, false, b[t].v, (short)0, acc[0][t], false, false);
      acc[1][t] = __builtin_amdgcn_wmma_f32_16x16x32_bf16(
          false, a1.v, false, b[t].v, (short)0, acc[1][t], false, false);
    }
    __syncthreads();
  }

  // ---- store: acc[g][t][r] -> row = row0+16g+r+8*hf, col = col0+16t+mr
  #pragma unroll
  for (int g = 0; g < 2; ++g) {
    #pragma unroll
    for (int t = 0; t < 8; ++t) {
      const int ccol = col0 + t * 16 + mr;
      #pragma unroll
      for (int r = 0; r < 8; ++r) {
        const size_t idx = (size_t)(row0 + g * 16 + r + 8 * hf) * ldc + ccol;
        if (Out)   Out[idx]   = acc[g][t][r];
        if (Outbf) Outbf[idx] = f2bf(acc[g][t][r]);
      }
    }
  }
}

// ---------------------------------------------------------------------------
// Host orchestration
// ---------------------------------------------------------------------------
static inline int cdiv(int a, int b) { return (a + b - 1) / b; }

extern "C" void kernel_launch(void* const* d_in, const int* in_sizes, int n_in,
                              void* d_out, int out_size, void* d_ws, size_t ws_size,
                              hipStream_t stream) {
  (void)in_sizes; (void)n_in; (void)out_size; (void)ws_size;
  const int N = NN, M = MM, H = HH, T = TT;
  const int H2 = 2 * H;

  // ---- inputs (setup_inputs order) ----
  const float* edge_rep  = (const float*)d_in[0];
  const float* cycle_rep = (const float*)d_in[1];
  const float* path_rep  = (const float*)d_in[2];
  const float* ec_ab_w   = (const float*)d_in[3];
  const float* ep_ab_w   = (const float*)d_in[4];   // [T, 2H, H]
  const float* pc_ab_w   = (const float*)d_in[5];
  const float* cp_ab_w   = (const float*)d_in[6];   // [T, 2H, H]
  const float* ee_w1   = (const float*)d_in[7];
  const float* ee_bn1  = (const float*)d_in[8];
  const float* ee_w2   = (const float*)d_in[9];
  const float* ee_bn2  = (const float*)d_in[10];
  const float* ecE_w1  = (const float*)d_in[11];
  const float* ecE_bn1 = (const float*)d_in[12];
  const float* ecE_w2  = (const float*)d_in[13];
  const float* ecE_bn2 = (const float*)d_in[14];
  const float* ecC_w1  = (const float*)d_in[15];
  const float* ecC_bn1 = (const float*)d_in[16];
  const float* ecC_w2  = (const float*)d_in[17];
  const float* ecC_bn2 = (const float*)d_in[18];
  const float* ecP_w1  = (const float*)d_in[19];
  const float* ecP_bn1 = (const float*)d_in[20];
  const float* ecP_w2  = (const float*)d_in[21];
  const float* ecP_bn2 = (const float*)d_in[22];
  const float* topM_w  = (const float*)d_in[23];
  const float* topM_bn = (const float*)d_in[24];
  const float* topC_w1 = (const float*)d_in[25];
  const float* topC_bn1= (const float*)d_in[26];
  const float* topC_w2 = (const float*)d_in[27];
  const float* topC_bn2= (const float*)d_in[28];
  const float* topP_w1 = (const float*)d_in[29];
  const float* topP_bn1= (const float*)d_in[30];
  const float* topP_w2 = (const float*)d_in[31];
  const float* topP_bn2= (const float*)d_in[32];
  const int* ee_map = (const int*)d_in[33];
  const int* ec_map = (const int*)d_in[34];
  const int* ce_map = (const int*)d_in[35];
  const int* ep_map = (const int*)d_in[36];   // [T, 2, M]
  const int* pe_map = (const int*)d_in[37];
  const int* cp_map = (const int*)d_in[38];
  const int* pc_map = (const int*)d_in[39];

  // ---- workspace allocator (f32 + bf16 regions, 16B aligned) ----
  char* base = (char*)d_ws;
  size_t off = 0;
  auto carve_f = [&](size_t n) { float* p = (float*)(base + off); off += n * 4; return p; };
  auto carve_b = [&](size_t n) { off = (off + 15) & ~(size_t)15;
                                 __bf16* p = (__bf16*)(base + off); off += n * 2; return p; };
  const size_t NHf = (size_t)N * H;

  // f32 scratch
  float* stats     = carve_f(1024);
  float* cnt       = carve_f(N);
  float* g1        = carve_f(2 * NHf);
  float* e2e       = carve_f(4 * NHf);
  float* h1        = carve_f(2 * NHf);
  float* ee_out    = carve_f(NHf);
  float* gbuf2     = carve_f(2 * NHf);
  float* gbuf4     = carve_f(4 * NHf);
  float* cycle_new = carve_f(2 * NHf);
  float* c2e       = carve_f(4 * NHf);
  float* path_new  = carve_f(8 * NHf);
  float* p2e       = carve_f(4 * NHf);
  float* ec_edge   = carve_f(NHf);
  float* ec_cycle  = carve_f(NHf);
  float* ec_path   = carve_f(4 * NHf);
  float* pc_g      = carve_f(2 * NHf);
  float* tmpH      = carve_f(NHf);

  // bf16 activation copies (WMMA A-operands)
  __bf16* edge_bf      = carve_b(NHf);
  __bf16* e2e_bf       = carve_b(4 * NHf);
  __bf16* h1_bf        = carve_b(2 * NHf);
  __bf16* ee_out_bf    = carve_b(NHf);
  __bf16* gbuf2_bf     = carve_b(2 * NHf);
  __bf16* cycle_new_bf = carve_b(2 * NHf);
  __bf16* c2e_bf       = carve_b(4 * NHf);
  __bf16* path_new_bf  = carve_b(8 * NHf);
  __bf16* p2e_bf       = carve_b(4 * NHf);
  __bf16* ec_edge_bf   = carve_b(NHf);
  __bf16* ec_cycle_bf  = carve_b(NHf);
  __bf16* ec_path_bf   = carve_b(4 * NHf);
  __bf16* cp_path_bf   = carve_b(4 * NHf);
  __bf16* cp_cycle_bf  = carve_b(NHf);
  __bf16* pc_g_bf      = carve_b(2 * NHf);

  float* out_edge  = (float*)d_out;
  float* out_cycle = out_edge + NHf;
  float* out_path  = out_cycle + NHf;

  // ---- helpers ----
  auto zero = [&](float* p, int n) {
    zero_f32_v4<<<cdiv(n / 4, 256), 256, 0, stream>>>((float4*)p, n / 4);
  };
  auto cvt = [&](const float* x, __bf16* y, size_t n) {
    cvt_bf16_v4<<<cdiv((int)(n / 4), 256), 256, 0, stream>>>((const float4*)x, y, (int)(n / 4));
  };
  auto prew = [&](const float* W, int Kin, int Kout) {   // f32 W -> bf16 W^T
    __bf16* Wt = carve_b((size_t)Kin * Kout);
    transpose_w_bf16<<<cdiv(Kin * Kout, 256), 256, 0, stream>>>(W, Wt, Kin, Kout);
    return Wt;
  };
  auto gather = [&](const float* src, int C, const int* map2M,
                    float* out, __bf16* outbf) {
    zero(out, N * 2 * C);
    zero(cnt, N);
    gather_count<<<cdiv(M, 256), 256, 0, stream>>>(map2M + M, cnt, M);
    gather_scatter_v4<<<cdiv(M * (C / 4), 256), 256, 0, stream>>>(
        src, map2M, map2M + M, out, M, C);
    gather_fin_v4<<<cdiv(N * (C / 4), 256), 256, 0, stream>>>(out, outbf, cnt, N, C);
  };
  auto gemm = [&](const __bf16* A0, int c0, const __bf16* A1, int c1,
                  const __bf16* A2, int c2, const __bf16* Wt, int Kin, int Kout,
                  float* Out, __bf16* Outbf, int ldc) {
    dim3 g(N / 256, Kout / 128);
    gemm_bf16_wmma<<<g, 256, 0, stream>>>(A0, c0, A1, c1, A2, c2, Wt, Kin, Kout,
                                          Out, Outbf, ldc);
  };
  auto bn = [&](float* x, const float* gb, int C,
                const float* res, float* y, __bf16* ybf) {
    zero(stats, 2 * C);
    dim3 gs(cdiv(C, 256), 64);
    bn_stats<<<gs, 256, 0, stream>>>(x, stats, N, C);
    bn_apply_relu_v4<<<cdiv((N * C) / 4, 256), 256, 0, stream>>>(
        x, stats, gb, y, ybf, res, N, C);
  };

  // ---- pre-transpose + convert all weights to bf16 (tiny, ~2.5 MB) ----
  __bf16* Wt_ee1  = prew(ee_w1,  5 * H, H2);
  __bf16* Wt_ee2  = prew(ee_w2,  H2, H);
  __bf16* Wt_ecE1 = prew(ecE_w1, 9 * H, H2);
  __bf16* Wt_ecE2 = prew(ecE_w2, H2, H);
  __bf16* Wt_ecC1 = prew(ecC_w1, H2, H2);
  __bf16* Wt_ecC2 = prew(ecC_w2, H2, H);
  __bf16* Wt_ecP1 = prew(ecP_w1, H2, H2);
  __bf16* Wt_ecP2 = prew(ecP_w2, H2, H);
  __bf16* Wt_topM = prew(topM_w, H2, H);
  __bf16* Wt_tC1  = prew(topC_w1, H2, H2);
  __bf16* Wt_tC2  = prew(topC_w2, H2, H);
  __bf16* Wt_tP1  = prew(topP_w1, H2, H2);
  __bf16* Wt_tP2  = prew(topP_w2, H2, H);
  __bf16* Wt_ecab = prew(ec_ab_w, H2, H);
  __bf16* Wt_pcab = prew(pc_ab_w, H2, H);
  __bf16* Wt_ep[TT]; __bf16* Wt_cp[TT];
  for (int t = 0; t < T; ++t) {
    Wt_ep[t] = prew(ep_ab_w + (size_t)t * H2 * H, H2, H);
    Wt_cp[t] = prew(cp_ab_w + (size_t)t * H2 * H, H2, H);
  }
  cvt(edge_rep, edge_bf, NHf);

  // ================= Edge_edge: two-hop gather + MLP2 =================
  gather(edge_rep, H, ee_map, g1, nullptr);                 // [N,2H]
  gather(g1, H2, ee_map, e2e, e2e_bf);                      // [N,4H]
  gemm(edge_bf, H, e2e_bf, 4 * H, nullptr, 0, Wt_ee1, 5 * H, H2, h1, nullptr, H2);
  bn(h1, ee_bn1, H2, nullptr, nullptr, h1_bf);
  gemm(h1_bf, H2, nullptr, 0, nullptr, 0, Wt_ee2, H2, H, ee_out, nullptr, H);
  bn(ee_out, ee_bn2, H, nullptr, nullptr, ee_out_bf);

  // ================= Edge_Cycle =================
  gather(edge_rep, H, ec_map, gbuf2, gbuf2_bf);
  gemm(gbuf2_bf, H2, nullptr, 0, nullptr, 0, Wt_ecab, H2, H, cycle_new, nullptr, H2);
  copy_cols_v4<<<cdiv(N * (H / 4), 256), 256, 0, stream>>>(cycle_new + H, H2,
                                                           cycle_rep, H, N, H);
  cvt(cycle_new, cycle_new_bf, 2 * NHf);
  gather(cycle_new, H2, ce_map, c2e, c2e_bf);               // [N,4H]
  for (int t = 0; t < T; ++t) {
    float*  pn_t  = path_new    + (size_t)t * 2 * NHf;
    __bf16* pnb_t = path_new_bf + (size_t)t * 2 * NHf;
    gather(edge_rep, H, ep_map + (size_t)t * 2 * M, gbuf2, gbuf2_bf);
    gemm(gbuf2_bf, H2, nullptr, 0, nullptr, 0, Wt_ep[t], H2, H, pn_t, nullptr, H2);
    copy_cols_v4<<<cdiv(N * (H / 4), 256), 256, 0, stream>>>(
        pn_t + H, H2, path_rep + (size_t)t * NHf, H, N, H);
    cvt(pn_t, pnb_t, 2 * NHf);
  }
  zero(p2e, (int)(4 * NHf));
  for (int t = 0; t < T; ++t) {
    gather(path_new + (size_t)t * 2 * NHf, H2, pe_map + (size_t)t * 2 * M, gbuf4, nullptr);
    accum_add_v4<<<cdiv((int)NHf, 256), 256, 0, stream>>>(
        (float4*)p2e, (const float4*)gbuf4, (int)NHf);
  }
  cvt(p2e, p2e_bf, 4 * NHf);
  // ec_edge = MLP2([edge | c2e | p2e])  (Kin = 9H)
  gemm(edge_bf, H, c2e_bf, 4 * H, p2e_bf, 4 * H, Wt_ecE1, 9 * H, H2, h1, nullptr, H2);
  bn(h1, ecE_bn1, H2, nullptr, nullptr, h1_bf);
  gemm(h1_bf, H2, nullptr, 0, nullptr, 0, Wt_ecE2, H2, H, ec_edge, nullptr, H);
  bn(ec_edge, ecE_bn2, H, nullptr, nullptr, ec_edge_bf);
  // ec_cycle = MLP2(cycle_new)
  gemm(cycle_new_bf, H2, nullptr, 0, nullptr, 0, Wt_ecC1, H2, H2, h1, nullptr, H2);
  bn(h1, ecC_bn1, H2, nullptr, nullptr, h1_bf);
  gemm(h1_bf, H2, nullptr, 0, nullptr, 0, Wt_ecC2, H2, H, ec_cycle, nullptr, H);
  bn(ec_cycle, ecC_bn2, H, nullptr, ec_cycle, ec_cycle_bf);   // f32 (gather src) + bf16
  // ec_path[t] = MLP2(path_new[t])  (per-t BN stats, matches vmap)
  for (int t = 0; t < T; ++t) {
    float*  ep_t  = ec_path    + (size_t)t * NHf;
    __bf16* epb_t = ec_path_bf + (size_t)t * NHf;
    gemm(path_new_bf + (size_t)t * 2 * NHf, H2, nullptr, 0, nullptr, 0,
         Wt_ecP1, H2, H2, h1, nullptr, H2);
    bn(h1, ecP_bn1, H2, nullptr, nullptr, h1_bf);
    gemm(h1_bf, H2, nullptr, 0, nullptr, 0, Wt_ecP2, H2, H, ep_t, nullptr, H);
    bn(ep_t, ecP_bn2, H, nullptr, ep_t, epb_t);               // f32 (gather src) + bf16
  }

  // ================= Cycle_path =================
  for (int t = 0; t < T; ++t) {
    gather(ec_cycle, H, cp_map + (size_t)t * 2 * M, gbuf2, gbuf2_bf);
    gemm(gbuf2_bf, H2, nullptr, 0, nullptr, 0, Wt_cp[t], H2, H,
         nullptr, cp_path_bf + (size_t)t * NHf, H);            // bf16-only output
  }
  zero(pc_g, (int)(2 * NHf));
  for (int t = 0; t < T; ++t) {
    gather(ec_path + (size_t)t * NHf, H, pc_map + (size_t)t * 2 * M, gbuf2, nullptr);
    accum_add_v4<<<cdiv((int)(NHf / 2), 256), 256, 0, stream>>>(
        (float4*)pc_g, (const float4*)gbuf2, (int)(NHf / 2));
  }
  cvt(pc_g, pc_g_bf, 2 * NHf);
  gemm(pc_g_bf, H2, nullptr, 0, nullptr, 0, Wt_pcab, H2, H, nullptr, cp_cycle_bf, H);

  // ================= top-level fuse + residual =================
  gemm(ee_out_bf, H, ec_edge_bf, H, nullptr, 0, Wt_topM, H2, H, tmpH, nullptr, H);
  bn(tmpH, topM_bn, H, edge_rep, out_edge, nullptr);
  gemm(cp_cycle_bf, H, ec_cycle_bf, H, nullptr, 0, Wt_tC1, H2, H2, h1, nullptr, H2);
  bn(h1, topC_bn1, H2, nullptr, nullptr, h1_bf);
  gemm(h1_bf, H2, nullptr, 0, nullptr, 0, Wt_tC2, H2, H, tmpH, nullptr, H);
  bn(tmpH, topC_bn2, H, cycle_rep, out_cycle, nullptr);
  for (int t = 0; t < T; ++t) {
    gemm(cp_path_bf + (size_t)t * NHf, H, ec_path_bf + (size_t)t * NHf, H, nullptr, 0,
         Wt_tP1, H2, H2, h1, nullptr, H2);
    bn(h1, topP_bn1, H2, nullptr, nullptr, h1_bf);
    gemm(h1_bf, H2, nullptr, 0, nullptr, 0, Wt_tP2, H2, H, tmpH, nullptr, H);
    bn(tmpH, topP_bn2, H, path_rep + (size_t)t * NHf,
       out_path + (size_t)t * NHf, nullptr);
  }
}
